// MoE_TopK_51488067944502
// MI455X (gfx1250) — matrix operand order repaired and checked by reading, compiled, not run on hardware
//
#include <hip/hip_runtime.h>
#include <hip/hip_bf16.h>
#include <math.h>

// ---------------------------------------------------------------------------
// MoE top-2 for MI455X (gfx1250, wave32).
//   B=4096 tokens, E=16 experts, D_IN=1024, D_HID=2048, D_OUT=1024.
// Routed (top-2) compute = 68.7 GFLOP; weights = 256MB f32 (~11us @ 23.3TB/s)
// -> roughly balanced vs bf16 WMMA ceiling, so: route first, read f32 weights
// once from HBM (per-expert 16MB slice stays L2-resident across token tiles),
// convert to bf16 in-register, and run both FFN GEMMs on
// v_wmma_f32_16x16x32_bf16 with f32 accumulation. H staged bf16 in ws (32MB).
// Each wave owns a 32x32 C patch (4 WMMA / K-step, A reused across 2 B cols).
// Out-of-range tile rows read a zeroed dummy row -> branch-free inner loop.
// ---------------------------------------------------------------------------

#define NTOK   4096
#define NEXP   16
#define DIN    1024
#define DHID   2048
#define DOUT   1024
#define RSLOTS (NTOK * 2)   // 8192 routed (token, expert) rows

typedef __bf16 bf16;
typedef __attribute__((ext_vector_type(16))) __bf16 v16bf;
typedef __attribute__((ext_vector_type(8)))  __bf16 v8bf;
typedef __attribute__((ext_vector_type(8)))  float  v8f;

__device__ __forceinline__ v8f wmma_bf16(v16bf a, v16bf b, v8f c) {
    // (neg_a, A, neg_b, B, c_mod, C, reuse_a, reuse_b)
    return __builtin_amdgcn_wmma_f32_16x16x32_bf16(false, a, false, b,
                                                   (short)0, c, false, false);
}

// A-fragment (16x32 bf16; M = lane&15, K-half = lane>>4) from an f32 row.
__device__ __forceinline__ v16bf load_a_f32(const float* __restrict__ row,
                                            int kbase, int khalf) {
    const float4* p0 = (const float4*)(row + kbase + khalf * 8);
    const float4* p1 = (const float4*)(row + kbase + 16 + khalf * 8);
    float4 u0 = p0[0], u1 = p0[1];
    float4 w0 = p1[0], w1 = p1[1];
    v16bf a;
    a[0]  = (bf16)u0.x; a[1]  = (bf16)u0.y; a[2]  = (bf16)u0.z; a[3]  = (bf16)u0.w;
    a[4]  = (bf16)u1.x; a[5]  = (bf16)u1.y; a[6]  = (bf16)u1.z; a[7]  = (bf16)u1.w;
    a[8]  = (bf16)w0.x; a[9]  = (bf16)w0.y; a[10] = (bf16)w0.z; a[11] = (bf16)w0.w;
    a[12] = (bf16)w1.x; a[13] = (bf16)w1.y; a[14] = (bf16)w1.z; a[15] = (bf16)w1.w;
    return a;
}

// A-fragment from an already-bf16 row (H scratch): two 16B loads.
__device__ __forceinline__ v16bf load_a_bf16(const bf16* __restrict__ row,
                                             int kbase, int khalf) {
    v8bf lo = *(const v8bf*)(row + kbase + khalf * 8);
    v8bf hi = *(const v8bf*)(row + kbase + 16 + khalf * 8);
    v16bf a;
#pragma unroll
    for (int j = 0; j < 8; ++j) { a[j] = lo[j]; a[8 + j] = hi[j]; }
    return a;
}

// B-fragment (32x16 bf16): lane col = lane&15, K = khalf*8 + [0,8) and
// 16 + khalf*8 + [0,8) (same as A). Weight matrix is [n][k] row-major -> K-contiguous.
__device__ __forceinline__ v16bf load_b_f32(const float* __restrict__ wrow,
                                            int kbase, int khalf) {
    const float4* p = (const float4*)(wrow + kbase + khalf * 8);
    const float4* pq = (const float4*)(wrow + kbase + 16 + khalf * 8);
    float4 q0 = p[0], q1 = p[1], q2 = pq[0], q3 = pq[1];
    v16bf b;
    b[0]  = (bf16)q0.x; b[1]  = (bf16)q0.y; b[2]  = (bf16)q0.z; b[3]  = (bf16)q0.w;
    b[4]  = (bf16)q1.x; b[5]  = (bf16)q1.y; b[6]  = (bf16)q1.z; b[7]  = (bf16)q1.w;
    b[8]  = (bf16)q2.x; b[9]  = (bf16)q2.y; b[10] = (bf16)q2.z; b[11] = (bf16)q2.w;
    b[12] = (bf16)q3.x; b[13] = (bf16)q3.y; b[14] = (bf16)q3.z; b[15] = (bf16)q3.w;
    return b;
}

// ---------------------------------------------------------------------------
// 0) zero output, per-expert counts, and the dummy zero rows
// ---------------------------------------------------------------------------
__global__ void moe_init_kernel(float* __restrict__ out, int n_out,
                                int* __restrict__ counts,
                                float* __restrict__ zerox,
                                bf16* __restrict__ zeroh) {
    int idx = blockIdx.x * blockDim.x + threadIdx.x;
    for (int i = idx; i < n_out; i += gridDim.x * blockDim.x) out[i] = 0.0f;
    if (idx < NEXP) counts[idx] = 0;
    if (idx < DIN)  zerox[idx] = 0.0f;
    if (idx < DHID) zeroh[idx] = (bf16)0.0f;
}

// ---------------------------------------------------------------------------
// 1) gate scores + top-2 + softmax. 256 threads: 16 tokens x 16 experts.
// ---------------------------------------------------------------------------
__global__ __launch_bounds__(256) void moe_gate_kernel(
    const float* __restrict__ x, const float* __restrict__ gate_w,
    int* __restrict__ tok_e, float* __restrict__ tok_w,
    int* __restrict__ counts) {
    __shared__ float sc[16][17];
    int t = threadIdx.x;
    int ltok = t >> 4, e = t & 15;
    int token = blockIdx.x * 16 + ltok;

    const float4* xr = (const float4*)(x + (size_t)token * DIN);
    const float4* gr = (const float4*)(gate_w + (size_t)e * DIN);
    float acc = 0.0f;
#pragma unroll 4
    for (int j = 0; j < DIN / 4; ++j) {
        float4 a = xr[j], b = gr[j];
        acc += a.x * b.x + a.y * b.y + a.z * b.z + a.w * b.w;
    }
    sc[ltok][e] = acc;
    __syncthreads();

    if (t < 16) {
        int token2 = blockIdx.x * 16 + t;
        float v0 = -INFINITY; int e0 = 0;
        for (int q = 0; q < NEXP; ++q)
            if (sc[t][q] > v0) { v0 = sc[t][q]; e0 = q; }
        float v1 = -INFINITY; int e1 = 0;
        for (int q = 0; q < NEXP; ++q)
            if (q != e0 && sc[t][q] > v1) { v1 = sc[t][q]; e1 = q; }
        float z  = __expf(v1 - v0);          // v0 >= v1
        float wA = 1.0f / (1.0f + z);
        float wB = z / (1.0f + z);
        tok_e[token2 * 2 + 0] = e0; tok_e[token2 * 2 + 1] = e1;
        tok_w[token2 * 2 + 0] = wA; tok_w[token2 * 2 + 1] = wB;
        atomicAdd(&counts[e0], 1);
        atomicAdd(&counts[e1], 1);
    }
}

// ---------------------------------------------------------------------------
// 2) deterministic routing: prefix-sum counts, then one thread per expert
//    scans all (token,k) entries in order and compacts its token list.
// ---------------------------------------------------------------------------
__global__ void moe_route_kernel(const int* __restrict__ counts,
                                 int* __restrict__ offsets,
                                 const int* __restrict__ tok_e,
                                 const float* __restrict__ tok_w,
                                 int* __restrict__ routed_token,
                                 float* __restrict__ routed_weight) {
    __shared__ int offs[NEXP + 1];
    if (threadIdx.x == 0) {
        int s = 0;
        for (int e = 0; e < NEXP; ++e) { offs[e] = s; s += counts[e]; }
        offs[NEXP] = s;
        for (int e = 0; e <= NEXP; ++e) offsets[e] = offs[e];
    }
    __syncthreads();
    int e = threadIdx.x;
    if (e < NEXP) {
        int slot = offs[e];
        for (int idx = 0; idx < RSLOTS; ++idx) {
            if (tok_e[idx] == e) {
                routed_token[slot]  = idx >> 1;
                routed_weight[slot] = tok_w[idx];
                ++slot;
            }
        }
    }
}

// ---------------------------------------------------------------------------
// 3) FFN1: H[row, n] = relu( x[tok(row)] . w1[e][n] + b1[e][n] ), bf16 out.
//    Grid: (expert, m-block of 64 rows, n-block of 128 hidden units).
//    8 waves = 2(M) x 4(N); each wave: 32x32 C patch = 4 WMMA accumulators.
// ---------------------------------------------------------------------------
__global__ __launch_bounds__(256) void moe_ffn1_kernel(
    const float* __restrict__ x,  const float* __restrict__ w1,
    const float* __restrict__ b1, const int* __restrict__ offsets,
    const int* __restrict__ routed_token, bf16* __restrict__ H,
    const float* __restrict__ zerox) {
    int e = blockIdx.x;
    int seg_lo = offsets[e], seg_hi = offsets[e + 1];
    int row0 = seg_lo + blockIdx.y * 64;
    if (row0 >= seg_hi) return;
    int n0 = blockIdx.z * 128;

    int tid = threadIdx.x, lane = tid & 31, wave = tid >> 5;
    int wm = wave & 1, wn = wave >> 1;
    int khalf = lane >> 4;
    int colA = n0 + wn * 32 + (lane & 15);
    int colB = colA + 16;

    int rb0 = row0 + wm * 32;            // first 16-row subtile
    int rb1 = rb0 + 16;                  // second 16-row subtile
    int ra  = rb0 + (lane & 15);
    int rbn = rb1 + (lane & 15);
    int tok0 = (ra  < seg_hi) ? routed_token[ra]  : -1;
    int tok1 = (rbn < seg_hi) ? routed_token[rbn] : -1;
    const float* xr0 = (tok0 >= 0) ? (x + (size_t)tok0 * DIN) : zerox;
    const float* xr1 = (tok1 >= 0) ? (x + (size_t)tok1 * DIN) : zerox;
    const float* wrowA = w1 + (size_t)e * DHID * DIN + (size_t)colA * DIN;
    const float* wrowB = w1 + (size_t)e * DHID * DIN + (size_t)colB * DIN;

    float biasA = b1[e * DHID + colA];
    float biasB = b1[e * DHID + colB];
    v8f c00, c01, c10, c11;
#pragma unroll
    for (int i = 0; i < 8; ++i) {
        c00[i] = biasA; c01[i] = biasB; c10[i] = biasA; c11[i] = biasB;
    }

    for (int k = 0; k < DIN; k += 32) {
        __builtin_prefetch(wrowA + k + 128, 0, 0);   // global_prefetch_b8
        __builtin_prefetch(wrowB + k + 128, 0, 0);
        v16bf a0 = load_a_f32(xr0, k, khalf);
        v16bf a1 = load_a_f32(xr1, k, khalf);
        v16bf bA = load_b_f32(wrowA, k, khalf);
        v16bf bB = load_b_f32(wrowB, k, khalf);
        c00 = wmma_bf16(a0, bA, c00);
        c01 = wmma_bf16(a0, bB, c01);
        c10 = wmma_bf16(a1, bA, c10);
        c11 = wmma_bf16(a1, bB, c11);
    }

    // C layout: VGPR r -> row r (lanes 0-15) / row 8+r (lanes 16-31), col=lane&15
    int rowOff = (lane >= 16) ? 8 : 0;
#pragma unroll
    for (int r = 0; r < 8; ++r) {
        int g0 = rb0 + r + rowOff;
        if (g0 < seg_hi) {
            H[(size_t)g0 * DHID + colA] = (bf16)fmaxf(c00[r], 0.0f);
            H[(size_t)g0 * DHID + colB] = (bf16)fmaxf(c01[r], 0.0f);
        }
        int g1 = rb1 + r + rowOff;
        if (g1 < seg_hi) {
            H[(size_t)g1 * DHID + colA] = (bf16)fmaxf(c10[r], 0.0f);
            H[(size_t)g1 * DHID + colB] = (bf16)fmaxf(c11[r], 0.0f);
        }
    }
}

// ---------------------------------------------------------------------------
// 4) FFN2: out[tok(row)] += gate_wt(row) * ( H[row] . w2[e][n] + b2[e][n] )
//    Each (token,col) receives exactly 2 atomic adds (commutative -> determ.)
// ---------------------------------------------------------------------------
__global__ __launch_bounds__(256) void moe_ffn2_kernel(
    const bf16* __restrict__ H,   const float* __restrict__ w2,
    const float* __restrict__ b2, const int* __restrict__ offsets,
    const int* __restrict__ routed_token,
    const float* __restrict__ routed_weight, float* __restrict__ out,
    const bf16* __restrict__ zeroh) {
    int e = blockIdx.x;
    int seg_lo = offsets[e], seg_hi = offsets[e + 1];
    int row0 = seg_lo + blockIdx.y * 64;
    if (row0 >= seg_hi) return;
    int n0 = blockIdx.z * 128;

    int tid = threadIdx.x, lane = tid & 31, wave = tid >> 5;
    int wm = wave & 1, wn = wave >> 1;
    int khalf = lane >> 4;
    int colA = n0 + wn * 32 + (lane & 15);
    int colB = colA + 16;

    int rb0 = row0 + wm * 32;
    int rb1 = rb0 + 16;
    int ra  = rb0 + (lane & 15);
    int rbn = rb1 + (lane & 15);
    const bf16* hr0 = (ra  < seg_hi) ? (H + (size_t)ra  * DHID) : zeroh;
    const bf16* hr1 = (rbn < seg_hi) ? (H + (size_t)rbn * DHID) : zeroh;
    const float* wrowA = w2 + (size_t)e * DOUT * DHID + (size_t)colA * DHID;
    const float* wrowB = w2 + (size_t)e * DOUT * DHID + (size_t)colB * DHID;

    float biasA = b2[e * DOUT + colA];
    float biasB = b2[e * DOUT + colB];
    v8f c00, c01, c10, c11;
#pragma unroll
    for (int i = 0; i < 8; ++i) {
        c00[i] = biasA; c01[i] = biasB; c10[i] = biasA; c11[i] = biasB;
    }

    for (int k = 0; k < DHID; k += 32) {
        __builtin_prefetch(wrowA + k + 128, 0, 0);
        __builtin_prefetch(wrowB + k + 128, 0, 0);
        v16bf a0 = load_a_bf16(hr0, k, khalf);
        v16bf a1 = load_a_bf16(hr1, k, khalf);
        v16bf bA = load_b_f32(wrowA, k, khalf);
        v16bf bB = load_b_f32(wrowB, k, khalf);
        c00 = wmma_bf16(a0, bA, c00);
        c01 = wmma_bf16(a0, bB, c01);
        c10 = wmma_bf16(a1, bA, c10);
        c11 = wmma_bf16(a1, bB, c11);
    }

    int rowOff = (lane >= 16) ? 8 : 0;
#pragma unroll
    for (int r = 0; r < 8; ++r) {
        int g0 = rb0 + r + rowOff;
        if (g0 < seg_hi) {
            int tok = routed_token[g0];
            float wgt = routed_weight[g0];
            unsafeAtomicAdd(&out[(size_t)tok * DOUT + colA], wgt * c00[r]);
            unsafeAtomicAdd(&out[(size_t)tok * DOUT + colB], wgt * c01[r]);
        }
        int g1 = rb1 + r + rowOff;
        if (g1 < seg_hi) {
            int tok = routed_token[g1];
            float wgt = routed_weight[g1];
            unsafeAtomicAdd(&out[(size_t)tok * DOUT + colA], wgt * c10[r]);
            unsafeAtomicAdd(&out[(size_t)tok * DOUT + colB], wgt * c11[r]);
        }
    }
}

// ---------------------------------------------------------------------------
extern "C" void kernel_launch(void* const* d_in, const int* in_sizes, int n_in,
                              void* d_out, int out_size, void* d_ws, size_t ws_size,
                              hipStream_t stream) {
    const float* x      = (const float*)d_in[0];
    const float* gate_w = (const float*)d_in[1];
    const float* w1     = (const float*)d_in[2];
    const float* b1     = (const float*)d_in[3];
    const float* w2     = (const float*)d_in[4];
    const float* b2     = (const float*)d_in[5];
    float* out = (float*)d_out;

    auto al = [](size_t v) { return (v + 255) & ~(size_t)255; };
    char* ws = (char*)d_ws;
    size_t off = 0;
    int*   counts        = (int*)(ws + off);   off = al(off + NEXP * sizeof(int));
    int*   offsets       = (int*)(ws + off);   off = al(off + (NEXP + 1) * sizeof(int));
    int*   tok_e         = (int*)(ws + off);   off = al(off + RSLOTS * sizeof(int));
    float* tok_w         = (float*)(ws + off); off = al(off + RSLOTS * sizeof(float));
    int*   routed_token  = (int*)(ws + off);   off = al(off + RSLOTS * sizeof(int));
    float* routed_weight = (float*)(ws + off); off = al(off + RSLOTS * sizeof(float));
    float* zerox         = (float*)(ws + off); off = al(off + DIN * sizeof(float));
    bf16*  zeroh         = (bf16*)(ws + off);  off = al(off + DHID * sizeof(bf16));
    bf16*  H             = (bf16*)(ws + off);  // RSLOTS * DHID * 2 bytes = 32MB

    moe_init_kernel<<<512, 256, 0, stream>>>(out, out_size, counts, zerox, zeroh);
    moe_gate_kernel<<<NTOK / 16, 256, 0, stream>>>(x, gate_w, tok_e, tok_w, counts);
    moe_route_kernel<<<1, 32, 0, stream>>>(counts, offsets, tok_e, tok_w,
                                           routed_token, routed_weight);
    moe_ffn1_kernel<<<dim3(NEXP, NTOK / 64, DHID / 128), 256, 0, stream>>>(
        x, w1, b1, offsets, routed_token, H, zerox);
    moe_ffn2_kernel<<<dim3(NEXP, NTOK / 64, DOUT / 128), 256, 0, stream>>>(
        H, w2, b2, offsets, routed_token, routed_weight, out, zeroh);
}
